// TFDonutSwinSelfAttention_58892591562943
// MI455X (gfx1250) — matrix operand order repaired and checked by reading, not compiled, not run on hardware
//
#include <hip/hip_runtime.h>
#include <hip/hip_bf16.h>

// ---------------------------------------------------------------------------
// Swin window self-attention for MI455X (gfx1250), bf16 WMMA + f32 accumulate.
// One workgroup (8 wave32) per window. All GEMMs via v_wmma_f32_16x16x32_bf16.
// Bank-conflict-padded LDS, vectorized staging, shuffle-parallel softmax.
// ---------------------------------------------------------------------------

#define WIN     7
#define SEQ     49          // 7*7 tokens per window
#define SPAD    64          // padded to 4 tiles of 16
#define DMODEL  128
#define NHEAD   4
#define HDIM    32
#define NWIN    64

// Padded LDS row strides (elements). All chosen so fragment loads stay
// 16-byte aligned while consecutive rows land on different banks.
#define LDA     136         // sX / sQ / sK rows (128 bf16 + 8 pad)
#define LDVT    72          // sVt / sP rows (64 bf16 + 8 pad)
#define LDSS    68          // sS rows (64 f32 + 4 pad)
#define LDW     144         // sW rows (128 bf16 + 16 pad)

// Byte offsets inside the shared-memory arena.
#define OFF_Q   0
#define OFF_K   (SPAD * LDA * 2)            // 17408
#define OFF_VT  (2 * SPAD * LDA * 2)        // 34816
#define OFF_X   (OFF_VT + DMODEL * LDVT * 2)// 53248 (sX phase1 / sS phase2)
#define OFF_W   (OFF_X + SPAD * LDA * 2)    // 70656 (sW phase1 / sP phase2)
#define SMEM_BYTES (OFF_W + DMODEL * LDW * 2) // 107520 (105 KB) -> 3 blocks/WGP

typedef __attribute__((ext_vector_type(16))) __bf16 v16bf;
typedef __attribute__((ext_vector_type(8)))  float  v8f;

union FragB16 {
    v16bf  v;
    uint4  u[2];
    __bf16 h[16];
};
union Pack8  { __bf16 h[8];  uint4 u;    };
union Pack4  { __bf16 h[4];  uint2 u;    };
union Pack16 { __bf16 h[16]; uint4 u[2]; };
union F4x4   { float4 f4[4]; float  f[16]; };

// A-matrix fragment, 16x32 bf16 tile from row-major [*, stride] buffer.
// ISA 7.12.2: lanes 0-15 -> M=lane, K = k0+{0..7} then k0+16+{0..7}
//             lanes 16-31 -> M=lane-16, K = k0+8+{0..7} then k0+24+{0..7}
__device__ inline v16bf load_a_frag(const __bf16* base, int stride,
                                    int m0, int k0, int lane) {
    FragB16 f;
    const __bf16* p = base + (m0 + (lane & 15)) * stride + k0 + ((lane >> 4) << 3);
    f.u[0] = *(const uint4*)p;
    f.u[1] = *(const uint4*)(p + 16);
    return f.v;
}

// B-matrix fragment, 32x16, where B[k][n] = base[n][k] (K contiguous per row).
// lanes 0-15: column N=lane, K=k0+{0..7} then k0+16+{0..7}; lanes 16-31: N=lane-16, K=k0+8+{0..7} then k0+24+{0..7}.
__device__ inline v16bf load_b_rows(const __bf16* base, int stride,
                                    int n0, int k0, int lane) {
    FragB16 f;
    const __bf16* p = base + (n0 + (lane & 15)) * stride + k0 + ((lane >> 4) << 3);
    f.u[0] = *(const uint4*)p;
    f.u[1] = *(const uint4*)(p + 16);
    return f.v;
}

__global__ __launch_bounds__(256)
void swin_window_attn_bf16(const float* __restrict__ X,       // [B,49,128]
                           const float* __restrict__ mask,    // [64,49,49]
                           const float* __restrict__ Wq, const float* __restrict__ bq,
                           const float* __restrict__ Wk, const float* __restrict__ bk,
                           const float* __restrict__ Wv, const float* __restrict__ bv,
                           const float* __restrict__ btab,    // [169,4]
                           float* __restrict__ out)           // [B,49,128]
{
    __shared__ __align__(16) unsigned char smem[SMEM_BYTES];
    __bf16* sQ  = (__bf16*)(smem + OFF_Q);   // [64][LDA]  row-major Q
    __bf16* sK  = (__bf16*)(smem + OFF_K);   // [64][LDA]  row-major K
    __bf16* sVt = (__bf16*)(smem + OFF_VT);  // [128][LDVT] V transposed: [dmodel][key]
    __bf16* sX  = (__bf16*)(smem + OFF_X);   // [64][LDA]  phase 1
    float*  sS  = (float*) (smem + OFF_X);   // [64][LDSS] phase 2 scores
    __bf16* sW  = (__bf16*)(smem + OFF_W);   // [128][LDW] phase 1 W^T
    __bf16* sP  = (__bf16*)(smem + OFF_W);   // [64][LDVT] phase 2 probs

    const int tid  = threadIdx.x;
    const int lane = tid & 31;
    const int wave = __builtin_amdgcn_readfirstlane(tid >> 5);   // wave-uniform SGPR

    const float* Xb = X + (size_t)blockIdx.x * SEQ * DMODEL;
    const float* mb = mask + (size_t)(blockIdx.x & (NWIN - 1)) * SEQ * SEQ;
    float* ob = out + (size_t)blockIdx.x * SEQ * DMODEL;

    // Warm the cache lines phase 2 will hit with scattered reads.
    if (wave == 0) {
        for (int i = lane; i < (SEQ * SEQ + 15) / 16; i += 32)
            __builtin_prefetch(mb + i * 16, 0, 3);
        if (lane < ((2 * WIN - 1) * (2 * WIN - 1) * NHEAD + 15) / 16)
            __builtin_prefetch(btab + lane * 16, 0, 3);
    }

    // ---- Phase 0: stage hidden_states -> bf16 LDS (vectorized), zero-pad ----
    for (int i = tid; i < SPAD * (DMODEL / 4); i += 256) {
        const int r  = i >> 5;          // row 0..63
        const int c4 = (i & 31) << 2;   // col 0,4,...,124
        float4 x = make_float4(0.f, 0.f, 0.f, 0.f);
        if (r < SEQ) x = *(const float4*)(Xb + r * DMODEL + c4);
        Pack4 pk;
        pk.h[0] = (__bf16)x.x; pk.h[1] = (__bf16)x.y;
        pk.h[2] = (__bf16)x.z; pk.h[3] = (__bf16)x.w;
        *(uint2*)(sX + r * LDA + c4) = pk.u;
    }
    __syncthreads();

    // ---- Phase 1: Q/K/V projections via WMMA (fully unrolled over p) -------
    const float* Ws[3] = { Wq, Wk, Wv };
    const float* bs[3] = { bq, bk, bv };
#pragma unroll
    for (int p = 0; p < 3; ++p) {
        const float* Wp = Ws[p];
        const float* bp = bs[p];

        // Stage W transposed: sW[n][k] = W[k][n]; 4 k-values packed per store.
        for (int i = tid; i < DMODEL * (DMODEL / 4); i += 256) {
            const int n  = i & 127;
            const int k0 = (i >> 7) << 2;
            Pack4 pk;
#pragma unroll
            for (int j = 0; j < 4; ++j)
                pk.h[j] = (__bf16)Wp[(k0 + j) * DMODEL + n];
            *(uint2*)(sW + n * LDW + k0) = pk.u;
        }
        __syncthreads();

        // 32 output tiles (4 Mtiles x 8 Ntiles); wave owns column nt = wave.
        const int n0 = wave * 16;
        const int n  = n0 + (lane & 15);
        const float bias = bp[n];
#pragma unroll
        for (int mt = 0; mt < 4; ++mt) {
            const int m0 = mt * 16;
            v8f acc = {};
#pragma unroll
            for (int kk = 0; kk < DMODEL; kk += 32) {
                v16bf a = load_a_frag(sX, LDA, m0, kk, lane);
                v16bf b = load_b_rows(sW, LDW, n0, kk, lane);
                acc = __builtin_amdgcn_wmma_f32_16x16x32_bf16(
                          false, a, false, b, (short)0, acc, false, false);
            }
            // D layout: lane l, vgpr v -> m = m0 + 8*(l>>4) + v, n = n0 + (l&15)
            const int mB = m0 + ((lane >> 4) << 3);
            if (p == 2) {
                // V: store transposed, 8 contiguous bf16 -> one b128 store.
                Pack8 pk;
#pragma unroll
                for (int v = 0; v < 8; ++v) pk.h[v] = (__bf16)(acc[v] + bias);
                *(uint4*)(sVt + n * LDVT + mB) = pk.u;
            } else {
                __bf16* dst = (p == 0) ? sQ : sK;
#pragma unroll
                for (int v = 0; v < 8; ++v)
                    dst[(mB + v) * LDA + n] = (__bf16)(acc[v] + bias);
            }
        }
        __syncthreads();
    }

    // ---- Phase 2: per-head attention ---------------------------------------
    const float scale = 0.17677669529663687f;   // 1/sqrt(32)
    for (int h = 0; h < NHEAD; ++h) {
        const int hk = h * HDIM;

        // Scores: Q_h [64x32] @ K_h^T [32x64] -> 16 tiles (4x4); 2 per wave.
#pragma unroll
        for (int i = 0; i < 2; ++i) {
            const int m0 = ((wave >> 2) + 2 * i) * 16;
            const int n0 = (wave & 3) * 16;
            v16bf a = load_a_frag(sQ, LDA, m0, hk, lane);
            v16bf b = load_b_rows(sK, LDA, n0, hk, lane);   // K^T: contiguous K
            v8f acc = {};
            acc = __builtin_amdgcn_wmma_f32_16x16x32_bf16(
                      false, a, false, b, (short)0, acc, false, false);

            const int n  = n0 + (lane & 15);
            const int mB = m0 + ((lane >> 4) << 3);
#pragma unroll
            for (int v = 0; v < 8; ++v) {
                const int m = mB + v;
                float sc = -1e30f;
                if (m < SEQ && n < SEQ) {
                    const int qi = m / WIN, qj = m - qi * WIN;
                    const int pi = n / WIN, pj = n - pi * WIN;
                    const int idx = (qi - pi + WIN - 1) * (2 * WIN - 1)
                                  + (qj - pj + WIN - 1);
                    sc = acc[v] * scale + btab[idx * NHEAD + h] + mb[m * SEQ + n];
                }
                sS[m * LDSS + n] = sc;
            }
        }
        __syncthreads();

        // Softmax: 4 threads per row, shuffle-reduced, bf16 probs -> sP.
        {
            const int row = tid >> 2;
            const int seg = (tid & 3) << 4;     // 0,16,32,48
            F4x4 e;
            const float4* r4 = (const float4*)(sS + row * LDSS + seg);
#pragma unroll
            for (int j = 0; j < 4; ++j) e.f4[j] = r4[j];
            float mx = -1e30f;
#pragma unroll
            for (int j = 0; j < 16; ++j) mx = fmaxf(mx, e.f[j]);
            mx = fmaxf(mx, __shfl_xor(mx, 1));
            mx = fmaxf(mx, __shfl_xor(mx, 2));
            float sum = 0.f;
#pragma unroll
            for (int j = 0; j < 16; ++j) { e.f[j] = __expf(e.f[j] - mx); sum += e.f[j]; }
            sum += __shfl_xor(sum, 1);
            sum += __shfl_xor(sum, 2);
            const float inv = 1.f / sum;
            Pack16 pk;
#pragma unroll
            for (int j = 0; j < 16; ++j) {
                const float pv = (row < SEQ && (seg + j) < SEQ) ? e.f[j] * inv : 0.f;
                pk.h[j] = (__bf16)pv;
            }
            uint4* dp = (uint4*)(sP + row * LDVT + seg);
            dp[0] = pk.u[0];
            dp[1] = pk.u[1];
        }
        __syncthreads();

        // Context: P [64x64] @ V_h [64x32] -> 8 tiles (4 Mtiles x 2 Ntiles).
        {
            const int m0 = (wave >> 1) * 16;
            const int n0 = (wave & 1) * 16;
            v8f acc = {};
#pragma unroll
            for (int kk = 0; kk < SPAD; kk += 32) {
                v16bf a = load_a_frag(sP, LDVT, m0, kk, lane);
                v16bf b = load_b_rows(sVt, LDVT, hk + n0, kk, lane); // V^T rows
                acc = __builtin_amdgcn_wmma_f32_16x16x32_bf16(
                          false, a, false, b, (short)0, acc, false, false);
            }
            const int n  = hk + n0 + (lane & 15);
            const int mB = m0 + ((lane >> 4) << 3);
#pragma unroll
            for (int v = 0; v < 8; ++v) {
                const int m = mB + v;
                if (m < SEQ) ob[m * DMODEL + n] = acc[v];
            }
        }
        __syncthreads();
    }
}

extern "C" void kernel_launch(void* const* d_in, const int* in_sizes, int n_in,
                              void* d_out, int out_size, void* d_ws, size_t ws_size,
                              hipStream_t stream) {
    const float* X    = (const float*)d_in[0];
    const float* mask = (const float*)d_in[1];
    const float* Wq   = (const float*)d_in[2];
    const float* bq   = (const float*)d_in[3];
    const float* Wk   = (const float*)d_in[4];
    const float* bk   = (const float*)d_in[5];
    const float* Wv   = (const float*)d_in[6];
    const float* bv   = (const float*)d_in[7];
    const float* btab = (const float*)d_in[8];
    float* out = (float*)d_out;

    const int B = in_sizes[0] / (SEQ * DMODEL);   // 4096 windows
    hipLaunchKernelGGL(swin_window_attn_bf16, dim3(B), dim3(256), 0, stream,
                       X, mask, Wq, bq, Wk, bk, Wv, bv, btab, out);
}
